// AdditiveAttention_41566693491048
// MI455X (gfx1250) — matrix operand order repaired and checked by reading, not compiled, not run on hardware
//
#include <hip/hip_runtime.h>

typedef __attribute__((ext_vector_type(16))) __bf16 v16bf;
typedef __attribute__((ext_vector_type(8)))  __bf16 v8bf;
typedef __attribute__((ext_vector_type(2)))  __bf16 v2bf;
typedef __attribute__((ext_vector_type(8)))  float  v8f;

#define HIDDEN 1024
#define BATCH  16
#define SEQ    2048
#define NEG_INF (-1000000000.0f)

#if __has_builtin(__builtin_amdgcn_tanhf)
  #define TANHF __builtin_amdgcn_tanhf
#elif __has_builtin(__builtin_amdgcn_tanh_f32)
  #define TANHF __builtin_amdgcn_tanh_f32
#else
  #define TANHF tanhf
#endif

// ---------------------------------------------------------------------------
// Kernel 1: q_proj[b][d] = sum_h query[b][h] * W1[h][d]     (16x1024, tiny)
// ---------------------------------------------------------------------------
__global__ __launch_bounds__(256)
void qproj_kernel(const float* __restrict__ query,
                  const float* __restrict__ W1,
                  float* __restrict__ qproj)
{
    const int b = blockIdx.x >> 2;
    const int d = ((blockIdx.x & 3) << 8) + threadIdx.x;
    float acc = 0.0f;
    for (int h = 0; h < HIDDEN; ++h)
        acc += query[b * HIDDEN + h] * W1[(size_t)h * HIDDEN + d];
    qproj[b * HIDDEN + d] = acc;
}

// ---------------------------------------------------------------------------
// Kernel 2: pack W2 (f32, HxH row-major) into bf16 WMMA-B-fragment layout.
// For (ktile 0..31, ntile 0..63): 32 lanes x 16 bf16, lane-major.
// Lane L: k = ktile*32 + (e>=8?16:0) + (L>=16?8:0) + e%8 (e=0..15), n = ntile*16 + L%16.
// ---------------------------------------------------------------------------
__global__ __launch_bounds__(256)
void packw2_kernel(const float* __restrict__ W2, __bf16* __restrict__ w2p)
{
    const int id = blockIdx.x * 256 + threadIdx.x;   // 0 .. 1048575
    const int e  = id & 15;
    const int L  = (id >> 4) & 31;
    const int nt = (id >> 9) & 63;
    const int kt = id >> 15;
    const int k  = kt * 32 + ((e >> 3) << 4) + ((L >> 4) << 3) + (e & 7);
    const int n  = nt * 16 + (L & 15);
    w2p[id] = (__bf16)W2[(size_t)k * HIDDEN + n];
}

// ---------------------------------------------------------------------------
// Kernel 3: fused k_proj GEMM (bf16 WMMA) + tanh + dot(v) -> scores.
// Workgroup = 32 rows x all 1024 cols. 8 waves; wave w owns cols [w*128,+128)
// as 8 n-tiles; two 16-row A fragments share every B fragment (2 WMMAs/load).
// grid: 1024 x 256
// ---------------------------------------------------------------------------
__global__ __launch_bounds__(256)
void scores_kernel(const float* __restrict__ keys,
                   const int*   __restrict__ mask,
                   const float* __restrict__ qproj,
                   const float* __restrict__ vvec,
                   const __bf16* __restrict__ w2p,
                   float* __restrict__ scores)
{
    __shared__ __bf16 lds_a[32][40];     // 32x32 A tile, padded stride (80B)
    __shared__ float  lds_p[8][32];      // per-wave row partials

    const int tid    = threadIdx.x;
    const int lane   = tid & 31;
    const int wid    = tid >> 5;          // wave 0..7
    const int r0     = blockIdx.x * 32;   // first flat row (b*SEQ + s)
    const int b      = r0 / SEQ;          // all 32 rows share one batch
    const int n_lane = lane & 15;
    const int hi     = lane >> 4;         // 0: lanes 0-15, 1: lanes 16-31

    // per-lane q_proj / v gathers (n = wid*128 + nt*16 + n_lane)
    float qpv[8], vvv[8];
#pragma unroll
    for (int nt = 0; nt < 8; ++nt) {
        const int n = wid * 128 + nt * 16 + n_lane;
        qpv[nt] = qproj[b * HIDDEN + n];
        vvv[nt] = vvec[n];
    }

    v8f acc0[8] = {};                     // C tiles for rows r0 .. r0+15
    v8f acc1[8] = {};                     // C tiles for rows r0+16 .. r0+31

    const int m   = n_lane;               // A-frag row within 16-row tile
    const int klo = hi * 8;               // A-frag k offset

    for (int kb = 0; kb < HIDDEN; kb += 32) {
        __syncthreads();
        // stage 32x32 keys tile as bf16: float2 load -> packed b32 LDS store
        {
            int q = tid;                  // pair index; 512 pairs total
#pragma unroll
            for (int i = 0; i < 2; ++i) {
                const int row = q >> 4;          // 0..31
                const int pc  = (q & 15) << 1;   // even col 0..30
                const float2 f = *(const float2*)
                    &keys[(size_t)(r0 + row) * HIDDEN + kb + pc];
                v2bf p; p.x = (__bf16)f.x; p.y = (__bf16)f.y;
                *(v2bf*)&lds_a[row][pc] = p;
                q += 256;
            }
        }
        __syncthreads();

        // A fragments: elems 0-7 = k klo..klo+7, elems 8-15 = k 16+klo..
        const v8bf a0lo = *(const v8bf*)&lds_a[m][klo];
        const v8bf a0hi = *(const v8bf*)&lds_a[m][16 + klo];
        const v16bf afrag0 = __builtin_shufflevector(
            a0lo, a0hi, 0, 1, 2, 3, 4, 5, 6, 7, 8, 9, 10, 11, 12, 13, 14, 15);
        const v8bf a1lo = *(const v8bf*)&lds_a[16 + m][klo];
        const v8bf a1hi = *(const v8bf*)&lds_a[16 + m][16 + klo];
        const v16bf afrag1 = __builtin_shufflevector(
            a1lo, a1hi, 0, 1, 2, 3, 4, 5, 6, 7, 8, 9, 10, 11, 12, 13, 14, 15);

        const int kt = kb >> 5;
#pragma unroll
        for (int nt = 0; nt < 8; ++nt) {
            const int ntg = wid * 8 + nt;
            const __bf16* bp =
                w2p + ((size_t)(kt * 64 + ntg) * 512) + (size_t)lane * 16;
            const v8bf b0 = *(const v8bf*)(bp);
            const v8bf b1 = *(const v8bf*)(bp + 8);
            const v16bf bfrag = __builtin_shufflevector(
                b0, b1, 0, 1, 2, 3, 4, 5, 6, 7, 8, 9, 10, 11, 12, 13, 14, 15);
            acc0[nt] = __builtin_amdgcn_wmma_f32_16x16x32_bf16(
                false, afrag0, false, bfrag, (short)0, acc0[nt], false, false);
            acc1[nt] = __builtin_amdgcn_wmma_f32_16x16x32_bf16(
                false, afrag1, false, bfrag, (short)0, acc1[nt], false, false);
        }
    }

    // tanh(q_proj + k_proj) * v, reduce over this wave's 128 columns.
    // C layout: lane -> n = n_lane; vgpr r -> row m = r + hi*8 (per 16-row tile)
    float sacc0[8], sacc1[8];
#pragma unroll
    for (int r = 0; r < 8; ++r) { sacc0[r] = 0.0f; sacc1[r] = 0.0f; }
#pragma unroll
    for (int nt = 0; nt < 8; ++nt) {
#pragma unroll
        for (int r = 0; r < 8; ++r) {
            sacc0[r] += TANHF(qpv[nt] + acc0[nt][r]) * vvv[nt];
            sacc1[r] += TANHF(qpv[nt] + acc1[nt][r]) * vvv[nt];
        }
    }
    // reduce across the 16 lanes of each half (xor 1,2,4,8 stays in-half)
#pragma unroll
    for (int r = 0; r < 8; ++r) {
#pragma unroll
        for (int off = 1; off < 16; off <<= 1) {
            sacc0[r] += __shfl_xor(sacc0[r], off, 32);
            sacc1[r] += __shfl_xor(sacc1[r], off, 32);
        }
    }
    if (n_lane == 0) {                    // lanes 0 and 16
#pragma unroll
        for (int r = 0; r < 8; ++r) {
            lds_p[wid][r + hi * 8]      = sacc0[r];
            lds_p[wid][16 + r + hi * 8] = sacc1[r];
        }
    }
    __syncthreads();
    if (tid < 32) {                       // ordered (deterministic) wave sum
        float s = 0.0f;
#pragma unroll
        for (int w = 0; w < 8; ++w) s += lds_p[w][tid];
        const int row = r0 + tid;
        scores[row] = (mask[row] == 0) ? NEG_INF : s;
    }
}

// ---------------------------------------------------------------------------
// Kernel 4: softmax over SEQ per batch. grid: 16 x 256 (8 elems/thread)
// ---------------------------------------------------------------------------
__global__ __launch_bounds__(256)
void softmax_kernel(const float* __restrict__ scores,
                    float* __restrict__ weights)
{
    __shared__ float red[256];
    const int b = blockIdx.x, tid = threadIdx.x;
    float vals[8];
    float mx = -3.4e38f;
#pragma unroll
    for (int i = 0; i < 8; ++i) {
        vals[i] = scores[b * SEQ + i * 256 + tid];
        mx = fmaxf(mx, vals[i]);
    }
    red[tid] = mx; __syncthreads();
    for (int s = 128; s > 0; s >>= 1) {
        if (tid < s) red[tid] = fmaxf(red[tid], red[tid + s]);
        __syncthreads();
    }
    mx = red[0]; __syncthreads();
    float sum = 0.0f;
#pragma unroll
    for (int i = 0; i < 8; ++i) { vals[i] = __expf(vals[i] - mx); sum += vals[i]; }
    red[tid] = sum; __syncthreads();
    for (int s = 128; s > 0; s >>= 1) {
        if (tid < s) red[tid] += red[tid + s];
        __syncthreads();
    }
    const float inv = 1.0f / red[0];
#pragma unroll
    for (int i = 0; i < 8; ++i)
        weights[b * SEQ + i * 256 + tid] = vals[i] * inv;
}

// ---------------------------------------------------------------------------
// Kernel 5: context partials: split SEQ into 8 chunks of 256. grid: 512 x 256
// ---------------------------------------------------------------------------
__global__ __launch_bounds__(256)
void ctx_partial_kernel(const float* __restrict__ values,
                        const float* __restrict__ weights,
                        float* __restrict__ part)
{
    const int b  = blockIdx.x & 15;
    const int sc = (blockIdx.x >> 4) & 7;
    const int hc = blockIdx.x >> 7;
    const int h  = hc * 256 + threadIdx.x;
    const int s0 = sc * 256;
    float acc = 0.0f;
    for (int s = s0; s < s0 + 256; ++s)
        acc += weights[b * SEQ + s] *
               values[((size_t)b * SEQ + s) * HIDDEN + h];
    part[((size_t)sc * BATCH + b) * HIDDEN + h] = acc;
}

// Kernel 6: ordered reduction of the 8 partials. grid: 64 x 256
__global__ __launch_bounds__(256)
void ctx_reduce_kernel(const float* __restrict__ part, float* __restrict__ ctx)
{
    const int i = blockIdx.x * 256 + threadIdx.x;   // 0..16383
    float acc = 0.0f;
#pragma unroll
    for (int sc = 0; sc < 8; ++sc) acc += part[sc * BATCH * HIDDEN + i];
    ctx[i] = acc;
}

// ---------------------------------------------------------------------------
extern "C" void kernel_launch(void* const* d_in, const int* in_sizes, int n_in,
                              void* d_out, int out_size, void* d_ws, size_t ws_size,
                              hipStream_t stream)
{
    (void)in_sizes; (void)n_in; (void)out_size; (void)ws_size;
    const float* query  = (const float*)d_in[0];
    const float* keys   = (const float*)d_in[1];
    const float* values = (const float*)d_in[2];
    const int*   mask   = (const int*)d_in[3];
    const float* W1     = (const float*)d_in[4];
    const float* W2     = (const float*)d_in[5];
    const float* v      = (const float*)d_in[6];

    float* ctx     = (float*)d_out;                       // (16,1024)
    float* weights = (float*)d_out + BATCH * HIDDEN;      // (16,2048)

    char* ws = (char*)d_ws;
    float*  qproj  = (float*)(ws);                                  // 64 KB
    __bf16* w2p    = (__bf16*)(ws + 65536);                         // 2 MB
    float*  scores = (float*)(ws + 65536 + 2097152);                // 128 KB
    float*  part   = (float*)(ws + 65536 + 2097152 + 131072);       // 512 KB

    qproj_kernel<<<64, 256, 0, stream>>>(query, W1, qproj);
    packw2_kernel<<<4096, 256, 0, stream>>>(W2, w2p);
    scores_kernel<<<(BATCH * SEQ) / 32, 256, 0, stream>>>(
        keys, mask, qproj, v, w2p, scores);
    softmax_kernel<<<BATCH, 256, 0, stream>>>(scores, weights);
    ctx_partial_kernel<<<512, 256, 0, stream>>>(values, weights, part);
    ctx_reduce_kernel<<<64, 256, 0, stream>>>(part, ctx);
}
